// TextGNN_47991964565536
// MI455X (gfx1250) — matrix operand order repaired and checked by reading, not compiled, not run on hardware
//
#include <hip/hip_runtime.h>
#include <hip/hip_bf16.h>

typedef __attribute__((ext_vector_type(16))) _Float16 v16h;
typedef __attribute__((ext_vector_type(8)))  float    v8f;

#define FEAT   128   // EMBED == HIDDEN == 128
#define NCLASS 20
#define NGRAPH 512

// ---------------------------------------------------------------------------
// utility: fill float buffer with a constant
__global__ void k_fill(float* __restrict__ p, float v, int n) {
    int i = blockIdx.x * blockDim.x + threadIdx.x;
    if (i < n) p[i] = v;
}

// degree accumulation over destination nodes (self-loop "1" pre-filled)
__global__ void k_deg_edges(const int* __restrict__ dst, float* __restrict__ deg, int E) {
    int i = blockIdx.x * blockDim.x + threadIdx.x;
    if (i < E) atomicAdd(&deg[dst[i]], 1.0f);
}

__global__ void k_rsqrt_inplace(float* __restrict__ d, int n) {
    int i = blockIdx.x * blockDim.x + threadIdx.x;
    if (i < n) d[i] = rsqrtf(d[i]);   // deg >= 1 always (self loop)
}

// embedding gather: feat[node][f] = emb[x[node]][f]
__global__ void k_embed(const int* __restrict__ x, const float* __restrict__ emb,
                        float* __restrict__ feat, int total) {
    int i = blockIdx.x * blockDim.x + threadIdx.x;
    if (i < total) {
        int node = i >> 7, f = i & 127;
        feat[i] = emb[(size_t)x[node] * FEAT + f];
    }
}

// convert 128x128 f32 weight to f16, transposed:  Wt[n][k] = W[k][n]
__global__ void k_wconv(const float* __restrict__ W, _Float16* __restrict__ Wt) {
    int i = blockIdx.x * blockDim.x + threadIdx.x;
    if (i < FEAT * FEAT) {
        int k = i >> 7, n = i & 127;
        Wt[n * FEAT + k] = (_Float16)W[k * FEAT + n];
    }
}

// ---------------------------------------------------------------------------
// WMMA GEMM: Y[N,128] = X[N,128] @ W[128,128]  (W given transposed f16 Wt[n][k])
// Block: 256 threads = 8 waves. Block handles 16 rows; wave w handles columns
// [16w, 16w+16). K=128 swept in four 16x16x32 WMMA steps, f32 accumulate.
__global__ void __launch_bounds__(256)
k_gemm(const float* __restrict__ X, const _Float16* __restrict__ Wt,
       float* __restrict__ Y, int N) {
    __shared__ _Float16 sX[16][FEAT];          // 4 KB  (A tile, f16)
    __shared__ _Float16 sWt[FEAT][FEAT];       // 32 KB (B, transposed [n][k])
    const int tid = threadIdx.x;
    const int r0  = blockIdx.x * 16;

    // stage whole transposed weight matrix (32 KB) with 16-byte copies
    {
        const uint4* src = (const uint4*)Wt;
        uint4*       dst = (uint4*)&sWt[0][0];
        for (int i = tid; i < (FEAT * FEAT * 2) / 16; i += 256) dst[i] = src[i];
    }
    // stage + convert the 16x128 X tile
    for (int i = tid; i < 16 * FEAT; i += 256) {
        int r = i >> 7, c = i & 127;
        float v = (r0 + r < N) ? X[(size_t)(r0 + r) * FEAT + c] : 0.0f;
        sX[r][c] = (_Float16)v;
    }
    __syncthreads();

    const int lane = tid & 31;
    const int wv   = tid >> 5;        // wave id = column tile 0..7
    const int half = lane >> 4;       // lane half (ISA layout)
    const int m    = lane & 15;
    const int n    = wv * 16 + m;     // this lane's B column / store column

    v8f acc = {};
    for (int kb = 0; kb < 4; ++kb) {
        v16h a, b;
#pragma unroll
        for (int j = 0; j < 16; ++j) {
            // A 16x32 f16 layout: lane-half selects K sub-block (ISA 7.12.2)
            int ka = kb * 32 + ((j < 8) ? j : j + 8) + 8 * half;
            a[j] = sX[m][ka];
            // B 32x16 f16 layout: same K order as A (lane-half selects K sub-block)
            int kbk = kb * 32 + ((j < 8) ? j : j + 8) + 8 * half;
            b[j] = sWt[n][kbk];
        }
        acc = __builtin_amdgcn_wmma_f32_16x16x32_f16(
                  false, a, false, b, (short)0, acc, false, false);
    }

    // D layout: VGPR r -> row (r + 8*half), col = lane%16.
    // Uniform branch: full tiles (the common case; N % 16 == 0 always hits
    // this) store unguarded; only a ragged last tile takes the guarded path.
    float* yb = Y + (size_t)(r0 + 8 * half) * FEAT + n;
    if (r0 + 16 <= N) {
#pragma unroll
        for (int r = 0; r < 8; ++r) yb[(size_t)r * FEAT] = acc[r];
    } else {
#pragma unroll
        for (int r = 0; r < 8; ++r) {
            if (r0 + r + 8 * half < N) yb[(size_t)r * FEAT] = acc[r];
        }
    }
}

// ---------------------------------------------------------------------------
// edge scatter: one wave per edge, lane handles 4 floats.
// agg[dst] += H[src] * dinv[src]*dinv[dst]   (atomic f32, L2-resident target)
__global__ void __launch_bounds__(256)
k_scatter(const float4* __restrict__ H, const float* __restrict__ dinv,
          const int* __restrict__ edges, float* __restrict__ agg, int E) {
    int gtid = blockIdx.x * 256 + threadIdx.x;
    int e    = gtid >> 5;
    if (e >= E) return;
    int lane = threadIdx.x & 31;
    int s = edges[e];
    int d = edges[E + e];
    float nrm = dinv[s] * dinv[d];
    float4 v = H[(size_t)s * (FEAT / 4) + lane];
    float* o = agg + (size_t)d * FEAT + lane * 4;
    atomicAdd(o + 0, v.x * nrm);
    atomicAdd(o + 1, v.y * nrm);
    atomicAdd(o + 2, v.z * nrm);
    atomicAdd(o + 3, v.w * nrm);
}

// epilogue: fold self-loop analytically, add bias, ReLU, write next features
__global__ void k_finish(const float* __restrict__ agg, const float* __restrict__ hw,
                         const float* __restrict__ dinv, const float* __restrict__ bias,
                         float* __restrict__ feat, int total) {
    int i = blockIdx.x * blockDim.x + threadIdx.x;
    if (i < total) {
        int node = i >> 7, f = i & 127;
        float di = dinv[node];
        float v  = agg[i] + hw[i] * di * di + bias[f];
        feat[i]  = v > 0.0f ? v : 0.0f;
    }
}

// per-graph pooling: wave per node, atomic add into pooled sums + counts
__global__ void __launch_bounds__(256)
k_pool(const float4* __restrict__ feat, const int* __restrict__ batch,
       float* __restrict__ pooled, float* __restrict__ cnts, int N) {
    int gtid = blockIdx.x * 256 + threadIdx.x;
    int node = gtid >> 5;
    if (node >= N) return;
    int lane = threadIdx.x & 31;
    int g = batch[node];
    float4 v = feat[(size_t)node * (FEAT / 4) + lane];
    float* o = pooled + (size_t)g * FEAT + lane * 4;
    atomicAdd(o + 0, v.x);
    atomicAdd(o + 1, v.y);
    atomicAdd(o + 2, v.z);
    atomicAdd(o + 3, v.w);
    if (lane == 0) atomicAdd(&cnts[g], 1.0f);
}

// classifier: out[g][c] = mean(pooled[g]) @ Wc + bc   (tiny: 512x20)
__global__ void k_classify(const float* __restrict__ pooled, const float* __restrict__ cnts,
                           const float* __restrict__ Wc, const float* __restrict__ bc,
                           float* __restrict__ out) {
    int g = blockIdx.x;
    int c = threadIdx.x;
    if (c >= NCLASS) return;
    float inv = 1.0f / fmaxf(cnts[g], 1.0f);
    const float* p = pooled + (size_t)g * FEAT;
    float s = 0.0f;
#pragma unroll 4
    for (int f = 0; f < FEAT; ++f) s += p[f] * Wc[f * NCLASS + c];
    out[g * NCLASS + c] = s * inv + bc[c];
}

// ---------------------------------------------------------------------------
extern "C" void kernel_launch(void* const* d_in, const int* in_sizes, int n_in,
                              void* d_out, int out_size, void* d_ws, size_t ws_size,
                              hipStream_t stream) {
    (void)n_in; (void)out_size; (void)ws_size;
    const int*   x     = (const int*)d_in[0];
    const int*   edges = (const int*)d_in[1];   // [2,E] flat: src then dst
    const int*   batch = (const int*)d_in[2];
    const float* emb   = (const float*)d_in[3];
    const float* W1    = (const float*)d_in[4];
    const float* b1    = (const float*)d_in[5];
    const float* W2    = (const float*)d_in[6];
    const float* b2    = (const float*)d_in[7];
    const float* Wc    = (const float*)d_in[8];
    const float* bc    = (const float*)d_in[9];
    float*       out   = (float*)d_out;

    const int N = in_sizes[0];
    const int E = in_sizes[1] / 2;
    const int NF = N * FEAT;

    // carve workspace (256B-aligned chunks)
    char* w = (char*)d_ws;
    auto carve = [&](size_t bytes) {
        void* p = (void*)w;
        w += (bytes + 255) & ~(size_t)255;
        return p;
    };
    float*    feat   = (float*)carve((size_t)NF * 4);
    float*    hw     = (float*)carve((size_t)NF * 4);
    float*    agg    = (float*)carve((size_t)NF * 4);
    float*    dinv   = (float*)carve((size_t)N * 4);
    _Float16* Wt1    = (_Float16*)carve(FEAT * FEAT * 2);
    _Float16* Wt2    = (_Float16*)carve(FEAT * FEAT * 2);
    float*    pooled = (float*)carve((size_t)NGRAPH * FEAT * 4);
    float*    cnts   = (float*)carve(NGRAPH * 4);

    const int T = 256;
    auto blks = [](int n, int t) { return (n + t - 1) / t; };

    // weights -> transposed f16 (once per call; cheap)
    k_wconv<<<blks(FEAT * FEAT, T), T, 0, stream>>>(W1, Wt1);
    k_wconv<<<blks(FEAT * FEAT, T), T, 0, stream>>>(W2, Wt2);

    // degrees with self-loops, then dinv = rsqrt(deg)
    k_fill<<<blks(N, T), T, 0, stream>>>(dinv, 1.0f, N);
    k_deg_edges<<<blks(E, T), T, 0, stream>>>(edges + E, dinv, E);
    k_rsqrt_inplace<<<blks(N, T), T, 0, stream>>>(dinv, N);

    // embedding lookup
    k_embed<<<blks(NF, T), T, 0, stream>>>(x, emb, feat, NF);

    const int gemm_grid    = blks(N, 16);
    const int scatter_grid = blks(E, 8);   // 8 waves per 256-thread block
    const int pool_grid    = blks(N, 8);

    // ---- layer 1 ----
    k_fill<<<blks(NF, T), T, 0, stream>>>(agg, 0.0f, NF);
    k_gemm<<<gemm_grid, T, 0, stream>>>(feat, Wt1, hw, N);
    k_scatter<<<scatter_grid, T, 0, stream>>>((const float4*)hw, dinv, edges, agg, E);
    k_finish<<<blks(NF, T), T, 0, stream>>>(agg, hw, dinv, b1, feat, NF);

    // ---- layer 2 ----
    k_fill<<<blks(NF, T), T, 0, stream>>>(agg, 0.0f, NF);
    k_gemm<<<gemm_grid, T, 0, stream>>>(feat, Wt2, hw, N);
    k_scatter<<<scatter_grid, T, 0, stream>>>((const float4*)hw, dinv, edges, agg, E);
    k_finish<<<blks(NF, T), T, 0, stream>>>(agg, hw, dinv, b2, feat, NF);

    // ---- pooling + classifier ----
    k_fill<<<blks(NGRAPH * FEAT, T), T, 0, stream>>>(pooled, 0.0f, NGRAPH * FEAT);
    k_fill<<<blks(NGRAPH, T), T, 0, stream>>>(cnts, 0.0f, NGRAPH);
    k_pool<<<pool_grid, T, 0, stream>>>((const float4*)feat, batch, pooled, cnts, N);
    k_classify<<<NGRAPH, 32, 0, stream>>>(pooled, cnts, Wc, bc, out);
}
